// Qwen3Attention_3384434229505
// MI455X (gfx1250) — matrix operand order repaired and checked by reading, not compiled, not run on hardware
//
#include <hip/hip_runtime.h>

#define S_LEN 2048
#define DIN 4096
#define NUM_HEADS 32
#define HEAD_DIM 128
#define NUM_KV 8
#define QDIM (NUM_HEADS * HEAD_DIM)   // 4096
#define KVDIM (NUM_KV * HEAD_DIM)     // 1024
#define QKVDIM (QDIM + 2 * KVDIM)     // 6144 (fused projection width)

typedef __bf16 bf16;
typedef __attribute__((ext_vector_type(16))) __bf16 v16bf;
typedef __attribute__((ext_vector_type(8)))  float  v8f;
typedef __attribute__((ext_vector_type(2))) __bf16 bf16x2;

union FragU { v16bf v; bf16x2 p[8]; };

__device__ inline v8f vzero8() {
  v8f z;
#pragma unroll
  for (int e = 0; e < 8; ++e) z[e] = 0.0f;
  return z;
}

// A-matrix (16x32 bf16, MxK) fragment: lane -> m = lane&15, hi = lane>>4.
// VGPR j holds K = 2j + hi*16 (same contiguous order as the B fragment).
__device__ inline v16bf load_a_frag(const bf16* __restrict__ A, int lda, int row0, int k0, int lane) {
  FragU f;
  int m = lane & 15, hi = lane >> 4;
  const bf16* rp = A + (size_t)(row0 + m) * lda + k0 + hi * 16;
#pragma unroll
  for (int j = 0; j < 8; ++j) {
    int kk = 2 * j;
    f.p[j] = *reinterpret_cast<const bf16x2*>(rp + kk);
  }
  return f.v;
}

// B-matrix (32x16 bf16, KxN) fragment from K-major (transposed) storage:
// element (k, n) lives at BT[(col0+n)*ldb + k0 + k]; lane n = lane&15, hi = lane>>4;
// VGPR j holds K = 2j + hi*16 (consecutive pairs -> one b32 load, coalesced to b128).
__device__ inline v16bf load_b_fragT(const bf16* __restrict__ BT, int ldb, int k0, int col0, int lane) {
  FragU f;
  int n = lane & 15, hi = lane >> 4;
  const bf16* cp = BT + (size_t)(col0 + n) * ldb + k0 + hi * 16;
#pragma unroll
  for (int j = 0; j < 8; ++j)
    f.p[j] = *reinterpret_cast<const bf16x2*>(cp + 2 * j);
  return f.v;
}

__device__ inline v8f wmma_bf16(v16bf a, v16bf b, v8f c) {
  return __builtin_amdgcn_wmma_f32_16x16x32_bf16(false, a, false, b, (short)0, c, false, false);
}

// ---------------------------------------------------------------- converts
__global__ void convert_f32_bf16(const float* __restrict__ src, bf16* __restrict__ dst, int nelem) {
  int stride = gridDim.x * blockDim.x;
  for (int i = blockIdx.x * blockDim.x + threadIdx.x; i < nelem; i += stride)
    dst[i] = (bf16)src[i];
}

// src: R x Ccols tile out of a row-major f32 matrix with leading dim src_ld
// -> dst: Ccols x R row-major bf16 (R, Ccols multiples of 32)
__global__ __launch_bounds__(256) void transpose_f32_bf16(const float* __restrict__ src,
                                                          bf16* __restrict__ dst,
                                                          int R, int Ccols, int src_ld) {
  __shared__ float tile[32][33];
  int tx = threadIdx.x & 31;
  int ty = threadIdx.x >> 5;                 // 0..7
  int br = blockIdx.y * 32;
  int bc = blockIdx.x * 32;
  for (int i = ty; i < 32; i += 8)
    tile[i][tx] = src[(size_t)(br + i) * src_ld + bc + tx];
  __syncthreads();
  for (int i = ty; i < 32; i += 8)
    dst[(size_t)(bc + i) * R + br + tx] = (bf16)tile[tx][i];
}

// ---------------------------------------------------------------- GEMM (bf16 in, f32 out)
// C[M,N] = A[M,K] (row-major bf16) * B[K,N], B supplied transposed as BT[N,K].
// Wave tile 64x64 (16 WMMAs per 8 fragment loads); 4 waves 2x2 -> block tile 128x128.
__global__ __launch_bounds__(128) void gemm_bf16(const bf16* __restrict__ A,
                                                 const bf16* __restrict__ BT,
                                                 float* __restrict__ C,
                                                 int M, int N, int K) {
  int lane = threadIdx.x & 31;
  int wave = threadIdx.x >> 5;
  int row0 = blockIdx.x * 128 + (wave >> 1) * 64;
  int col0 = blockIdx.y * 128 + (wave & 1) * 64;
  v8f acc[4][4];
#pragma unroll
  for (int i = 0; i < 4; ++i)
#pragma unroll
    for (int j = 0; j < 4; ++j) acc[i][j] = vzero8();

  int pl = lane & 15;
  for (int k0 = 0; k0 < K; k0 += 32) {
    if (k0 + 256 < K) {  // speculative RT prefetch of a future k-tile (global_prefetch_b8)
      __builtin_prefetch(A + (size_t)(row0 + pl) * K + k0 + 256, 0, 3);
      __builtin_prefetch(BT + (size_t)(col0 + pl) * K + k0 + 256, 0, 3);
    }
    v16bf a[4];
#pragma unroll
    for (int mi = 0; mi < 4; ++mi) a[mi] = load_a_frag(A, K, row0 + mi * 16, k0, lane);
#pragma unroll
    for (int nt = 0; nt < 4; ++nt) {
      v16bf b = load_b_fragT(BT, K, k0, col0 + nt * 16, lane);
#pragma unroll
      for (int mi = 0; mi < 4; ++mi) acc[mi][nt] = wmma_bf16(a[mi], b, acc[mi][nt]);
    }
  }
  int n = lane & 15, hi = lane >> 4;
#pragma unroll
  for (int mi = 0; mi < 4; ++mi)
#pragma unroll
    for (int nt = 0; nt < 4; ++nt)
#pragma unroll
      for (int r = 0; r < 8; ++r)
        C[(size_t)(row0 + mi * 16 + hi * 8 + r) * N + col0 + nt * 16 + n] = acc[mi][nt][r];
}

// ---------------------------------------------------------------- RMSNorm + RoPE (f32 -> bf16)
// One wave per (token, head); 128 elements = 4 per lane; rotate-half partner is in-lane.
__global__ __launch_bounds__(128) void normrope_kernel(const float* __restrict__ in, int in_ld,
                                                       const int* __restrict__ positions,
                                                       const float* __restrict__ scale,
                                                       bf16* __restrict__ out, int nheads) {
  int lane = threadIdx.x & 31;
  int w = blockIdx.x * 4 + (threadIdx.x >> 5);
  int token = w / nheads;
  int h = w - token * nheads;
  const float* row = in + (size_t)token * in_ld + h * HEAD_DIM;
  float v[4], ss = 0.f;
#pragma unroll
  for (int i = 0; i < 4; ++i) { v[i] = row[lane + 32 * i]; ss += v[i] * v[i]; }
#pragma unroll
  for (int off = 16; off >= 1; off >>= 1) ss += __shfl_xor(ss, off, 32);
  float inv = rsqrtf(ss * (1.0f / HEAD_DIM) + 1e-6f);
  float xn[4];
#pragma unroll
  for (int i = 0; i < 4; ++i) xn[i] = v[i] * inv * scale[lane + 32 * i];
  float pos = (float)positions[token];
  bf16* orow = out + (size_t)token * nheads * HEAD_DIM + h * HEAD_DIM;
#pragma unroll
  for (int i = 0; i < 4; ++i) {
    int d = lane + 32 * i;
    int j = d & 63;                                        // freq index
    float ang = pos * expf((float)j * -0.14391156514261f); // -ln(10000)/64
    float s, c;
    sincosf(ang, &s, &c);
    float rot;
    if (i < 2) rot = -xn[i + 2];                           // d < 64: -x[d+64]
    else       rot =  xn[(i + 2) & 3];                     // d >= 64: x[d-64]
    orow[d] = (bf16)(xn[i] * c + rot * s);
  }
}

// ---------------------------------------------------------------- Flash attention (causal, GQA)
// One wave per (q-tile of 16 rows, head). Per 32-key block: 8 WMMAs for S,
// online softmax in C-layout registers, P transposed C->A layout via LDS,
// 8 WMMAs for P*V against token-major-transposed V.
__global__ __launch_bounds__(128) void attn_kernel(const bf16* __restrict__ Q,
                                                   const bf16* __restrict__ Kb,
                                                   const bf16* __restrict__ Vt,
                                                   bf16* __restrict__ ctx) {
  __shared__ bf16 pstage[4][16 * 32];
  int lane = threadIdx.x & 31;
  int wv = threadIdx.x >> 5;
  int w = blockIdx.x * 4 + wv;
  int qt = w >> 5;            // 0..127
  int head = w & 31;          // 0..31
  int kv = head >> 2;         // GROUP = 4
  int qrow0 = qt * 16;
  int n = lane & 15, hi = lane >> 4;

  v16bf qa[4];
#pragma unroll
  for (int dt = 0; dt < 4; ++dt)
    qa[dt] = load_a_frag(Q, QDIM, qrow0, head * HEAD_DIM + dt * 32, lane);

  v8f cacc[8];
#pragma unroll
  for (int dt = 0; dt < 8; ++dt) cacc[dt] = vzero8();
  float mrow[8], lrow[8];
#pragma unroll
  for (int r = 0; r < 8; ++r) { mrow[r] = -1e30f; lrow[r] = 0.f; }
  const float sc = 0.08838834764831845f;  // 1/sqrt(128)
  bf16* sm = &pstage[wv][0];
  const bf16* vbase = Vt + (size_t)kv * HEAD_DIM * S_LEN;

  int kend = qrow0 + 16;
  for (int kb = 0; kb < kend; kb += 32) {
    v8f sacc[2];
    sacc[0] = vzero8();
    sacc[1] = vzero8();
#pragma unroll
    for (int t = 0; t < 2; ++t)
#pragma unroll
      for (int dt = 0; dt < 4; ++dt) {
        // B(k=d, n=key): K rows are contiguous over d -> K-major fragment load
        v16bf bk = load_b_fragT(Kb, KVDIM, kv * HEAD_DIM + dt * 32, kb + t * 16, lane);
        sacc[t] = wmma_bf16(qa[dt], bk, sacc[t]);
      }
    float p[2][8];
    bool needmask = (kb + 31 > qrow0);
#pragma unroll
    for (int t = 0; t < 2; ++t)
#pragma unroll
      for (int r = 0; r < 8; ++r) {
        float val = sacc[t][r] * sc;
        if (needmask && (kb + t * 16 + n > qrow0 + hi * 8 + r)) val = -1e30f;
        p[t][r] = val;
      }
    // row max over 32 cols: local pair max then 16-lane half-wave reduction
    float rmax[8];
#pragma unroll
    for (int r = 0; r < 8; ++r) rmax[r] = fmaxf(p[0][r], p[1][r]);
#pragma unroll
    for (int off = 1; off < 16; off <<= 1)
#pragma unroll
      for (int r = 0; r < 8; ++r) rmax[r] = fmaxf(rmax[r], __shfl_xor(rmax[r], off, 32));
    float corr[8], rsum[8];
#pragma unroll
    for (int r = 0; r < 8; ++r) {
      float mn = fmaxf(mrow[r], rmax[r]);
      corr[r] = expf(mrow[r] - mn);
      mrow[r] = mn;
      p[0][r] = expf(p[0][r] - mn);
      p[1][r] = expf(p[1][r] - mn);
      rsum[r] = p[0][r] + p[1][r];
    }
#pragma unroll
    for (int off = 1; off < 16; off <<= 1)
#pragma unroll
      for (int r = 0; r < 8; ++r) rsum[r] += __shfl_xor(rsum[r], off, 32);
#pragma unroll
    for (int r = 0; r < 8; ++r) lrow[r] = lrow[r] * corr[r] + rsum[r];
#pragma unroll
    for (int dt = 0; dt < 8; ++dt)
#pragma unroll
      for (int r = 0; r < 8; ++r) cacc[dt][r] *= corr[r];

    // P: C-layout -> A-layout transpose through LDS (wave-private 1KB)
#pragma unroll
    for (int t = 0; t < 2; ++t)
#pragma unroll
      for (int r = 0; r < 8; ++r)
        sm[(hi * 8 + r) * 32 + t * 16 + n] = (bf16)p[t][r];
    asm volatile("s_wait_dscnt 0" ::: "memory");   // DS ops are in-order per wave; fence the compiler too
    v16bf pa = load_a_frag(sm, 32, 0, 0, lane);

#pragma unroll
    for (int dt = 0; dt < 8; ++dt) {
      v16bf bv = load_b_fragT(vbase, S_LEN, kb, dt * 16, lane);
      cacc[dt] = wmma_bf16(pa, bv, cacc[dt]);
    }
  }
  float invl[8];
#pragma unroll
  for (int r = 0; r < 8; ++r) invl[r] = 1.0f / lrow[r];
#pragma unroll
  for (int dt = 0; dt < 8; ++dt)
#pragma unroll
    for (int r = 0; r < 8; ++r)
      ctx[(size_t)(qrow0 + hi * 8 + r) * QDIM + head * HEAD_DIM + dt * 16 + n] =
          (bf16)(cacc[dt][r] * invl[r]);
}

// ---------------------------------------------------------------- launcher
extern "C" void kernel_launch(void* const* d_in, const int* in_sizes, int n_in,
                              void* d_out, int out_size, void* d_ws, size_t ws_size,
                              hipStream_t stream) {
  (void)in_sizes; (void)n_in; (void)out_size; (void)ws_size;
  const float* x   = (const float*)d_in[0];
  const int*   pos = (const int*)d_in[1];
  const float* Wq  = (const float*)d_in[2];
  const float* Wk  = (const float*)d_in[3];
  const float* Wv  = (const float*)d_in[4];
  const float* Wo  = (const float*)d_in[5];
  const float* qs  = (const float*)d_in[6];
  const float* ks  = (const float*)d_in[7];
  float* out = (float*)d_out;

  char* ws = (char*)d_ws;
  size_t off = 0;
  bf16*  x16    = (bf16*)(ws + off);  off += (size_t)S_LEN * DIN * 2;      // 16 MB
  bf16*  wqkvT  = (bf16*)(ws + off);  off += (size_t)QKVDIM * DIN * 2;     // 48 MB (fused, N x K)
  bf16*  woT    = (bf16*)(ws + off);  off += (size_t)DIN * QDIM * 2;       // 32 MB
  float* qkvf   = (float*)(ws + off); off += (size_t)S_LEN * QKVDIM * 4;   // 48 MB (reused as ctx16)
  bf16*  q16    = (bf16*)(ws + off);  off += (size_t)S_LEN * QDIM * 2;     // 16 MB
  bf16*  k16    = (bf16*)(ws + off);  off += (size_t)S_LEN * KVDIM * 2;    //  4 MB
  bf16*  vt16   = (bf16*)(ws + off);  off += (size_t)KVDIM * S_LEN * 2;    //  4 MB
  bf16*  ctx16  = (bf16*)qkvf;  // qkvf fully consumed before attention writes ctx

  // 1) precision conversion (once): x -> bf16; Wq|Wk|Wv packed transposed bf16; Wo transposed
  convert_f32_bf16<<<4096, 256, 0, stream>>>(x, x16, S_LEN * DIN);
  transpose_f32_bf16<<<dim3(QDIM / 32, DIN / 32), 256, 0, stream>>>(
      Wq, wqkvT, DIN, QDIM, QDIM);
  transpose_f32_bf16<<<dim3(KVDIM / 32, DIN / 32), 256, 0, stream>>>(
      Wk, wqkvT + (size_t)QDIM * DIN, DIN, KVDIM, KVDIM);
  transpose_f32_bf16<<<dim3(KVDIM / 32, DIN / 32), 256, 0, stream>>>(
      Wv, wqkvT + (size_t)(QDIM + KVDIM) * DIN, DIN, KVDIM, KVDIM);
  transpose_f32_bf16<<<dim3(DIN / 32, QDIM / 32), 256, 0, stream>>>(
      Wo, woT, QDIM, DIN, DIN);

  // 2) fused QKV projection (WMMA): [2048 x 4096] @ [4096 x 6144]
  gemm_bf16<<<dim3(S_LEN / 128, QKVDIM / 128), 128, 0, stream>>>(
      x16, wqkvT, qkvf, S_LEN, QKVDIM, DIN);

  // 3) QK RMSNorm + RoPE -> bf16 ; V -> token-major-transposed bf16
  normrope_kernel<<<(S_LEN * NUM_HEADS) / 4, 128, 0, stream>>>(
      qkvf, QKVDIM, pos, qs, q16, NUM_HEADS);
  normrope_kernel<<<(S_LEN * NUM_KV) / 4, 128, 0, stream>>>(
      qkvf + QDIM, QKVDIM, pos, ks, k16, NUM_KV);
  transpose_f32_bf16<<<dim3(KVDIM / 32, S_LEN / 32), 256, 0, stream>>>(
      qkvf + QDIM + KVDIM, vt16, S_LEN, KVDIM, QKVDIM);

  // 4) causal flash attention (WMMA for S and P*V)
  attn_kernel<<<(NUM_HEADS * (S_LEN / 16)) / 4, 128, 0, stream>>>(q16, k16, vt16, ctx16);

  // 5) output projection (WMMA) -> f32 out
  gemm_bf16<<<dim3(S_LEN / 128, DIN / 128), 128, 0, stream>>>(ctx16, woT, out, S_LEN, DIN, QDIM);
}
